// GraphLayer_75187697483851
// MI455X (gfx1250) — matrix operand order repaired and checked by reading, not compiled, not run on hardware
//
#include <hip/hip_runtime.h>
#include <hip/hip_bf16.h>

#define HIDDEN 512
#define LN_EPS 1e-5f

typedef __bf16 v16bf __attribute__((ext_vector_type(16)));
typedef float  v8f   __attribute__((ext_vector_type(8)));

__device__ __forceinline__ unsigned short f32_to_bf16(float f) {
    unsigned int u = __float_as_uint(f);
    unsigned int r = (u + 0x7FFFu + ((u >> 16) & 1u)) >> 16;   // RNE
    return (unsigned short)r;
}

// CDNA5 async copy: global memory -> LDS, per-lane 16B, tracked by ASYNCcnt.
__device__ __forceinline__ void async_copy_b128(void* lds, const void* g) {
    asm volatile("global_load_async_to_lds_b128 %0, %1, off"
                 :: "v"((unsigned)(unsigned long long)lds), "v"(g)
                 : "memory");
}
__device__ __forceinline__ void wait_async0() {
    asm volatile("s_wait_asynccnt 0x0" ::: "memory");
}

// ---------------- degree / dinv ----------------
__global__ void k_deg_init(float* deg, int n) {
    int i = blockIdx.x * blockDim.x + threadIdx.x;
    if (i < n) deg[i] = 1.0f;                 // self-loop contributes 1
}

__global__ void k_deg_count(const long long* __restrict__ ei, float* deg,
                            int nedges) {
    int e = blockIdx.x * blockDim.x + threadIdx.x;
    if (e < nedges) {
        int d = (int)ei[(long long)nedges + e];   // dst row
        atomicAdd(&deg[d], 1.0f);
    }
}

__global__ void k_deg_finalize(float* deg, int n) {
    int i = blockIdx.x * blockDim.x + threadIdx.x;
    if (i < n) deg[i] = rsqrtf(deg[i]);           // deg >= 1 always
}

// ---------------- f32 -> bf16 casts ----------------
__global__ void k_cast_bf16(const float* __restrict__ src,
                            unsigned short* __restrict__ dst, long long n) {
    long long i = (long long)blockIdx.x * blockDim.x + threadIdx.x;
    long long stride = (long long)gridDim.x * blockDim.x;
    for (; i < n; i += stride) dst[i] = f32_to_bf16(src[i]);
}

// W [512,512] f32 -> Wt [n][k] bf16 (transpose so GEMM B-tiles are contiguous)
__global__ void k_cast_w_t(const float* __restrict__ w,
                           unsigned short* __restrict__ wt) {
    int idx = blockIdx.x * blockDim.x + threadIdx.x;
    if (idx < HIDDEN * HIDDEN) {
        int k = idx >> 9, n = idx & 511;
        wt[n * HIDDEN + k] = f32_to_bf16(w[idx]);
    }
}

// ---------------- WMMA GEMM: h = X @ W ----------------
// Block tile 128(M) x 64(N), K-step 32, double-buffered async LDS staging.
// 8 waves; each wave owns a 16x64 strip (4 accumulators).
#define TM 128
#define TN 64
#define TK 32

__global__ __launch_bounds__(256) void k_gemm_bf16(
    const unsigned short* __restrict__ A,    // [nrows,512] bf16 (X)
    const unsigned short* __restrict__ Bt,   // [512,512]  bf16 (W transposed, [n][k])
    float* __restrict__ C, int nrows)
{
    __shared__ unsigned short sA[2][TM][TK];   // 2 x 8 KB
    __shared__ unsigned short sBt[2][TN][TK];  // 2 x 4 KB

    const int tid  = threadIdx.x;
    const int wave = tid >> 5;
    const int lane = tid & 31;
    const int hl   = lane >> 4;          // lane half (0/1)
    const int ln16 = lane & 15;

    const int rowBase = blockIdx.x * TM;
    const int colBase = blockIdx.y * TN;

    // per-thread fixed staging chunks (16B each); uniform, no divergence
    // A tile: 512 chunks (128 rows x 4); this thread owns tid and tid+256
    const int ar0 = tid >> 2,          ac0 = tid & 3;
    const int ar1 = (tid + 256) >> 2,  ac1 = (tid + 256) & 3;
    // B tile: 256 chunks (64 n-rows x 4); this thread owns tid
    const int bn = tid >> 2, bc = tid & 3;

    int gra0 = rowBase + ar0; if (gra0 >= nrows) gra0 = nrows - 1;
    int gra1 = rowBase + ar1; if (gra1 >= nrows) gra1 = nrows - 1;

    v8f acc[4];
    #pragma unroll
    for (int t = 0; t < 4; ++t) acc[t] = (v8f){0.f,0.f,0.f,0.f,0.f,0.f,0.f,0.f};

    auto stage = [&](int buf, int k0) {
        async_copy_b128(&sA[buf][ar0][ac0 * 8],
                        A + (long long)gra0 * HIDDEN + k0 + ac0 * 8);
        async_copy_b128(&sA[buf][ar1][ac1 * 8],
                        A + (long long)gra1 * HIDDEN + k0 + ac1 * 8);
        async_copy_b128(&sBt[buf][bn][bc * 8],
                        Bt + (long long)(colBase + bn) * HIDDEN + k0 + bc * 8);
    };

    stage(0, 0);
    wait_async0();
    __syncthreads();

    int buf = 0;
    for (int k0 = 0; k0 < HIDDEN; k0 += TK) {
        if (k0 + TK < HIDDEN) stage(buf ^ 1, k0 + TK);   // prefetch next tile

        // A fragment (16x32): lane(hl,ln16) holds row ln16, K per ISA layout
        union { unsigned int u[8]; v16bf v; } af;
        #pragma unroll
        for (int v = 0; v < 8; ++v) {
            int kb = 2 * v + (v >= 4 ? 8 : 0) + 8 * hl;
            af.u[v] = *(const unsigned int*)&sA[buf][wave * 16 + ln16][kb];
        }

        #pragma unroll
        for (int t = 0; t < 4; ++t) {
            // B fragment (32x16): n = t*16+ln16, K per ISA layout (same as A)
            union { unsigned int u[8]; v16bf v; } bfr;
            #pragma unroll
            for (int v = 0; v < 8; ++v)
                bfr.u[v] = *(const unsigned int*)
                           &sBt[buf][t * 16 + ln16][2 * v + (v >= 4 ? 8 : 0) + 8 * hl];

            acc[t] = __builtin_amdgcn_wmma_f32_16x16x32_bf16(
                false, af.v, false, bfr.v, (short)0, acc[t], false, false);
        }

        wait_async0();        // staged tile (buf^1) fully in LDS
        __syncthreads();      // all waves done reading buf / writing buf^1
        buf ^= 1;
    }

    // D layout: element v -> row (v + 8*hl), col ln16
    #pragma unroll
    for (int t = 0; t < 4; ++t) {
        #pragma unroll
        for (int v = 0; v < 8; ++v) {
            int gm = rowBase + wave * 16 + v + 8 * hl;
            if (gm < nrows)
                C[(long long)gm * HIDDEN + colBase + t * 16 + ln16] = acc[t][v];
        }
    }
}

// ---------------- out = residual + b + dinv^2 * h  (self-loop term) ----------------
__global__ void k_init_out(const float* __restrict__ x, const float* __restrict__ h,
                           const float* __restrict__ dinv, const float* __restrict__ b,
                           float* __restrict__ out, long long total) {
    long long idx = (long long)blockIdx.x * blockDim.x + threadIdx.x;
    long long stride = (long long)gridDim.x * blockDim.x;
    for (; idx < total; idx += stride) {
        int i = (int)(idx >> 9);          // /512
        int c = (int)(idx & 511);
        float di = dinv[i];
        out[idx] = x[idx] + b[c] + di * di * h[idx];
    }
}

// ---------------- per-edge gather + scatter-add ----------------
// one wave32 per edge; 512 floats = 128 float4 per row
__global__ __launch_bounds__(256) void k_edge_scatter(
    const long long* __restrict__ ei, const float* __restrict__ h,
    const float* __restrict__ dinv, float* __restrict__ out, int nedges)
{
    int wid  = threadIdx.x >> 5;
    int lane = threadIdx.x & 31;
    int e = blockIdx.x * 8 + wid;
    if (e >= nedges) return;

    int s = (int)ei[e];
    int d = (int)ei[(long long)nedges + e];
    float nrm = dinv[s] * dinv[d];

    const float4* hs = (const float4*)(h + (long long)s * HIDDEN);
    float* od = out + (long long)d * HIDDEN;
    __builtin_prefetch(hs, 0, 3);          // global_prefetch_b8

    #pragma unroll
    for (int i = 0; i < 4; ++i) {
        int c = lane + i * 32;             // float4 chunk index 0..127
        float4 v = hs[c];
        atomicAdd(&od[c * 4 + 0], v.x * nrm);
        atomicAdd(&od[c * 4 + 1], v.y * nrm);
        atomicAdd(&od[c * 4 + 2], v.z * nrm);
        atomicAdd(&od[c * 4 + 3], v.w * nrm);
    }
}

// ---------------- LayerNorm (in place), one wave per row ----------------
__global__ __launch_bounds__(256) void k_layernorm(
    float* __restrict__ out, const float* __restrict__ gamma,
    const float* __restrict__ beta, int nrows)
{
    int wid  = threadIdx.x >> 5;
    int lane = threadIdx.x & 31;
    int row = blockIdx.x * 8 + wid;
    if (row >= nrows) return;

    float* p = out + (long long)row * HIDDEN;
    float4 vals[4];
    float s = 0.f, s2 = 0.f;
    #pragma unroll
    for (int i = 0; i < 4; ++i) {
        float4 v = ((const float4*)p)[lane + i * 32];
        vals[i] = v;
        s  += v.x + v.y + v.z + v.w;
        s2 += v.x * v.x + v.y * v.y + v.z * v.z + v.w * v.w;
    }
    #pragma unroll
    for (int off = 16; off > 0; off >>= 1) {
        s  += __shfl_xor(s,  off, 32);
        s2 += __shfl_xor(s2, off, 32);
    }
    float mean = s * (1.0f / HIDDEN);
    float var  = s2 * (1.0f / HIDDEN) - mean * mean;
    float inv  = rsqrtf(var + LN_EPS);

    #pragma unroll
    for (int i = 0; i < 4; ++i) {
        int ci = lane + i * 32;
        float4 g = ((const float4*)gamma)[ci];
        float4 bt = ((const float4*)beta)[ci];
        float4 v = vals[i];
        v.x = (v.x - mean) * inv * g.x + bt.x;
        v.y = (v.y - mean) * inv * g.y + bt.y;
        v.z = (v.z - mean) * inv * g.z + bt.z;
        v.w = (v.w - mean) * inv * g.w + bt.w;
        ((float4*)p)[ci] = v;
    }
}

extern "C" void kernel_launch(void* const* d_in, const int* in_sizes, int n_in,
                              void* d_out, int out_size, void* d_ws, size_t ws_size,
                              hipStream_t stream) {
    const float*     X     = (const float*)d_in[0];      // [N,512]
    const long long* ei    = (const long long*)d_in[1];  // [2,E] int64
    const float*     W     = (const float*)d_in[2];      // [512,512]
    const float*     b     = (const float*)d_in[3];
    const float*     gamma = (const float*)d_in[4];
    const float*     beta  = (const float*)d_in[5];
    float*           out   = (float*)d_out;

    const int N = in_sizes[0] / HIDDEN;
    const int E = in_sizes[1] / 2;
    const long long NH = (long long)N * HIDDEN;
    const long long WW = (long long)HIDDEN * HIDDEN;

    // workspace carve-up (all 16B-aligned given sizes)
    unsigned short* Xbf  = (unsigned short*)d_ws;        // N*H bf16
    unsigned short* Wtbf = Xbf + NH;                     // H*H bf16 (transposed)
    float*          h    = (float*)(Wtbf + WW);          // N*H f32
    float*          dinv = h + NH;                       // N   f32 (deg then dinv)

    // 1) degree -> dinv
    k_deg_init<<<(N + 255) / 256, 256, 0, stream>>>(dinv, N);
    k_deg_count<<<(E + 255) / 256, 256, 0, stream>>>(ei, dinv, E);
    k_deg_finalize<<<(N + 255) / 256, 256, 0, stream>>>(dinv, N);

    // 2) bf16 copies: X as-is, W transposed
    k_cast_bf16<<<4096, 256, 0, stream>>>(X, Xbf, NH);
    k_cast_w_t<<<(int)((WW + 255) / 256), 256, 0, stream>>>(W, Wtbf);

    // 3) h = X @ W  (WMMA bf16 -> f32, async double-buffered staging)
    dim3 ggrid((N + TM - 1) / TM, HIDDEN / TN);
    k_gemm_bf16<<<ggrid, 256, 0, stream>>>(Xbf, Wtbf, h, N);

    // 4) out = residual + b + self-loop term
    k_init_out<<<4096, 256, 0, stream>>>(X, h, dinv, b, out, NH);

    // 5) scatter messages over edges (wave per edge)
    k_edge_scatter<<<(E + 7) / 8, 256, 0, stream>>>(ei, h, dinv, out, E);

    // 6) LayerNorm in place
    k_layernorm<<<(N + 7) / 8, 256, 0, stream>>>(out, gamma, beta, N);
}
